// DICE_12180527252014
// MI455X (gfx1250) — matrix operand order repaired and checked by reading, not compiled, not run on hardware
//
#include <hip/hip_runtime.h>
#include <hip/hip_bf16.h>
#include <math.h>

#define N_NODESC 50000
#define N_EDGESC 400000
#define HID      128
#define N_GRAPHS 64
#define DEPTH    3
#define BN_EPS   1e-5f

typedef __attribute__((ext_vector_type(16))) __bf16 v16bf;
typedef __attribute__((ext_vector_type(8)))  __bf16 bf16x8;
typedef __attribute__((ext_vector_type(4)))  __bf16 bf16x4;
typedef __attribute__((ext_vector_type(8)))  float  v8f;
typedef __attribute__((ext_vector_type(4)))  float  f32x4;

__device__ __forceinline__ __bf16 f2bf(float f) {
  unsigned int u = __float_as_uint(f);
  u += 0x7FFFu + ((u >> 16) & 1u);               // round-to-nearest-even
  unsigned short h = (unsigned short)(u >> 16);
  return __builtin_bit_cast(__bf16, h);
}

__device__ __forceinline__ float gelu_exact(float x) {
  return 0.5f * x * (1.0f + erff(x * 0.70710678118654752440f));
}

// ---------------------------------------------------------------------------
// WMMA GEMM: Y[rows,128] = act(X[rows,128] @ W[128,128] + b)
// One block = 128 rows (8 waves x 16-row M tiles). W staged transposed in LDS
// as bf16 so B fragments are contiguous ds_load_b128; X tile staged as bf16.
// Bias staged in LDS so accumulator init needs no mid-loop global load.
// ---------------------------------------------------------------------------
__global__ __launch_bounds__(256) void dice_wmma_gemm128(
    const float* __restrict__ X, int rows,
    const float* __restrict__ W, const float* __restrict__ bias,
    float* __restrict__ Y, int act)
{
  __shared__ __bf16 Xs[128][136];   // [row][k]   (pad 136 -> 272B rows, 16B aligned)
  __shared__ __bf16 Wt[128][136];   // [n][k] = W[k][n]
  __shared__ float  bsh[128];
  const int tid = threadIdx.x;
  const int rowBase = blockIdx.x * 128;

  // W: 4096 float4 vectors, transposed scatter into LDS
  for (int i = tid; i < 4096; i += 256) {
    int idx = i * 4;
    int k = idx >> 7, n = idx & 127;
    f32x4 wv = *(const f32x4*)&W[idx];
#pragma unroll
    for (int j = 0; j < 4; ++j) Wt[n + j][k] = f2bf(wv[j]);
  }
  // X tile: 4096 float4 vectors -> bf16x4 LDS stores (ds_store_b64)
  for (int i = tid; i < 4096; i += 256) {
    int idx = i * 4;
    int r = idx >> 7, c = idx & 127;
    int g = rowBase + r;
    bf16x4 pk;
    if (g < rows) {
      f32x4 xv = *(const f32x4*)&X[(size_t)g * HID + c];
#pragma unroll
      for (int j = 0; j < 4; ++j) pk[j] = f2bf(xv[j]);
    } else {
#pragma unroll
      for (int j = 0; j < 4; ++j) pk[j] = f2bf(0.0f);
    }
    *(bf16x4*)&Xs[r][c] = pk;
  }
  if (tid < 128) bsh[tid] = bias[tid];
  __syncthreads();

  const int wave = tid >> 5;
  const int lane = tid & 31;
  const int m    = lane & 15;
  const int half = lane >> 4;
  const int rloc = wave * 16 + m;

  // A fragments: lane l holds row m=l&15; element j -> K = kt*32 + (j<8?0:16) + half*8 + (j&7)
  v16bf afrag[4];
#pragma unroll
  for (int kt = 0; kt < 4; ++kt) {
    int k0 = kt * 32 + half * 8;
    bf16x8 lo = *(const bf16x8*)&Xs[rloc][k0];
    bf16x8 hi = *(const bf16x8*)&Xs[rloc][k0 + 16];
    afrag[kt] = __builtin_shufflevector(lo, hi, 0,1,2,3,4,5,6,7,8,9,10,11,12,13,14,15);
  }

#pragma unroll
  for (int nt = 0; nt < 8; ++nt) {
    int n = nt * 16 + m;
    float bv = bsh[n];
    v8f acc = { bv, bv, bv, bv, bv, bv, bv, bv };   // C rows share column n
#pragma unroll
    for (int kt = 0; kt < 4; ++kt) {
      // B fragment: lane l holds col n=l&15; element j -> K = kt*32 + (j<8?0:16) + half*8 + (j&7)
      int k0 = kt * 32 + half * 8;
      bf16x8 blo = *(const bf16x8*)&Wt[n][k0];
      bf16x8 bhi = *(const bf16x8*)&Wt[n][k0 + 16];
      v16bf bfrag = __builtin_shufflevector(blo, bhi, 0,1,2,3,4,5,6,7,8,9,10,11,12,13,14,15);
      acc = __builtin_amdgcn_wmma_f32_16x16x32_bf16(
          false, afrag[kt], false, bfrag, (short)0, acc, false, false);
    }
#pragma unroll
    for (int r = 0; r < 8; ++r) {
      int grow = rowBase + wave * 16 + half * 8 + r;
      if (grow < rows) {
        float v = acc[r];
        if (act) v = gelu_exact(v);
        Y[(size_t)grow * HID + n] = v;
      }
    }
  }
}

// ---------------------------------------------------------------------------
// Small-din input layer: h[r][c..c+3] = gelu(x[r,:] @ W[:,c..c+3] + b)
// One thread -> 4 output columns; x row loaded once.
// ---------------------------------------------------------------------------
__global__ void dice_embed_l1(const float* __restrict__ x, int rows, int din,
                              const float* __restrict__ W, const float* __restrict__ b,
                              float* __restrict__ h)
{
  __shared__ float Ws[9 * 128];
  __shared__ float bs[128];
  for (int i = threadIdx.x; i < din * 128; i += blockDim.x) Ws[i] = W[i];
  if (threadIdx.x < 128) bs[threadIdx.x] = b[threadIdx.x];
  __syncthreads();
  size_t total = (size_t)rows * 32;                 // vec4 groups
  size_t stride = (size_t)gridDim.x * blockDim.x;
  for (size_t v = (size_t)blockIdx.x * blockDim.x + threadIdx.x; v < total; v += stride) {
    size_t r = v >> 5;
    int c = (int)(v & 31) * 4;
    float xrow[9];
    for (int d = 0; d < din; ++d) xrow[d] = x[r * din + d];
    f32x4 acc = *(const f32x4*)&bs[c];
    for (int d = 0; d < din; ++d) acc += xrow[d] * *(const f32x4*)&Ws[d * 128 + c];
#pragma unroll
    for (int j = 0; j < 4; ++j) acc[j] = gelu_exact(acc[j]);
    *(f32x4*)&h[r * HID + c] = acc;
  }
}

// ---------------------------------------------------------------------------
// BatchNorm (training mode) helpers
// ---------------------------------------------------------------------------
__global__ void dice_fill4(float* __restrict__ p, size_t n4, float v) {
  // n4 = element count / 4 (all buffers are multiples of 4 floats)
  f32x4 vv = { v, v, v, v };
  size_t stride = (size_t)gridDim.x * blockDim.x;
  for (size_t i = (size_t)blockIdx.x * blockDim.x + threadIdx.x; i < n4; i += stride)
    *(f32x4*)&p[i * 4] = vv;
}

__global__ void dice_bn_stats(const float* __restrict__ x, int rows, float* __restrict__ sums) {
  __shared__ float ls[2 * HID];
  if (threadIdx.x < 2 * HID) ls[threadIdx.x] = 0.0f;
  __syncthreads();
  size_t total = (size_t)rows * 32;
  size_t stride = (size_t)gridDim.x * blockDim.x;
  for (size_t v = (size_t)blockIdx.x * blockDim.x + threadIdx.x; v < total; v += stride) {
    int c = (int)(v & 31) * 4;
    f32x4 xv = *(const f32x4*)&x[v * 4];
#pragma unroll
    for (int j = 0; j < 4; ++j) {
      atomicAdd(&ls[c + j], xv[j]);
      atomicAdd(&ls[c + j + HID], xv[j] * xv[j]);
    }
  }
  __syncthreads();
  if (threadIdx.x < 2 * HID) atomicAdd(&sums[threadIdx.x], ls[threadIdx.x]);
}

__global__ void dice_bn_finalize(const float* __restrict__ sums,
                                 const float* __restrict__ g, const float* __restrict__ b,
                                 float inv_rows, float* __restrict__ scale, float* __restrict__ shift) {
  int c = threadIdx.x;
  float mean = sums[c] * inv_rows;
  float var  = sums[c + HID] * inv_rows - mean * mean;
  float s = g[c] * rsqrtf(var + BN_EPS);
  scale[c] = s;
  shift[c] = b[c] - mean * s;
}

__global__ void dice_bn_apply(float* __restrict__ x, int rows,
                              const float* __restrict__ scale, const float* __restrict__ shift) {
  __shared__ float sc[128], sh[128];
  if (threadIdx.x < 128) { sc[threadIdx.x] = scale[threadIdx.x]; sh[threadIdx.x] = shift[threadIdx.x]; }
  __syncthreads();
  size_t total = (size_t)rows * 32;
  size_t stride = (size_t)gridDim.x * blockDim.x;
  for (size_t v = (size_t)blockIdx.x * blockDim.x + threadIdx.x; v < total; v += stride) {
    int c = (int)(v & 31) * 4;
    f32x4 xv = *(const f32x4*)&x[v * 4];
    f32x4 s  = *(const f32x4*)&sc[c];
    f32x4 t  = *(const f32x4*)&sh[c];
    *(f32x4*)&x[v * 4] = xv * s + t;
  }
}

// ---------------------------------------------------------------------------
// Attention / message passing
// ---------------------------------------------------------------------------
__global__ void dice_attn_logit(const float* __restrict__ nh, const float* __restrict__ eh,
                                const int* __restrict__ src, const int* __restrict__ dst,
                                float* __restrict__ logit, int E)
{
  int wid  = (int)((blockIdx.x * blockDim.x + threadIdx.x) >> 5);
  int lane = threadIdx.x & 31;
  int nw   = (int)((gridDim.x * blockDim.x) >> 5);
  for (int e = wid; e < E; e += nw) {
    const float* ps = nh + (size_t)src[e] * HID + lane * 4;
    const float* pd = nh + (size_t)dst[e] * HID + lane * 4;
    const float* pe = eh + (size_t)e * HID + lane * 4;
    f32x4 a = *(const f32x4*)ps;
    f32x4 b = *(const f32x4*)pe;
    f32x4 d = *(const f32x4*)pd;
    f32x4 m = (a + b) * d;
    float acc = m[0] + m[1] + m[2] + m[3];
#pragma unroll
    for (int k = 16; k >= 1; k >>= 1) acc += __shfl_xor(acc, k, 32);
    if (lane == 0) logit[e] = acc;
  }
}

__device__ __forceinline__ void atomicMaxF(float* addr, float val) {
  if (val >= 0.0f) atomicMax((int*)addr, __float_as_int(val));
  else             atomicMin((unsigned int*)addr, __float_as_uint(val));
}

__global__ void dice_seg_max(const float* __restrict__ logit, const int* __restrict__ dst,
                             float* __restrict__ mx, int E) {
  int stride = gridDim.x * blockDim.x;
  for (int e = blockIdx.x * blockDim.x + threadIdx.x; e < E; e += stride)
    atomicMaxF(&mx[dst[e]], logit[e]);
}

__global__ void dice_seg_expsum(const float* __restrict__ logit, const int* __restrict__ dst,
                                const float* __restrict__ mx, float* __restrict__ ex,
                                float* __restrict__ ssum, int E) {
  int stride = gridDim.x * blockDim.x;
  for (int e = blockIdx.x * blockDim.x + threadIdx.x; e < E; e += stride) {
    int d = dst[e];
    float v = __expf(logit[e] - mx[d]);
    ex[e] = v;
    atomicAdd(&ssum[d], v);
  }
}

__global__ void dice_nz_scatter(const float* __restrict__ nh, const int* __restrict__ src,
                                const int* __restrict__ dst, const float* __restrict__ ex,
                                const float* __restrict__ ssum, float* __restrict__ nz, int E)
{
  long long total = (long long)E * 32;
  long long stride = (long long)gridDim.x * blockDim.x;
  for (long long t = (long long)blockIdx.x * blockDim.x + threadIdx.x; t < total; t += stride) {
    int e = (int)(t >> 5);
    int lane = (int)(t & 31);
    int d = dst[e];
    float coef = ex[e] / ssum[d];
    f32x4 sv = *(const f32x4*)(nh + (size_t)src[e] * HID + lane * 4);
    float* pn = nz + (size_t)d * HID + lane * 4;
#pragma unroll
    for (int j = 0; j < 4; ++j) atomicAdd(&pn[j], sv[j] * coef);
  }
}

__global__ void dice_combine_node(const float* __restrict__ nh, const float* __restrict__ nz,
                                  const float* __restrict__ eps, float* __restrict__ xn, int rows) {
  size_t total = (size_t)rows * 32;
  size_t stride = (size_t)gridDim.x * blockDim.x;
  for (size_t v = (size_t)blockIdx.x * blockDim.x + threadIdx.x; v < total; v += stride) {
    int c = (int)(v & 31) * 4;
    f32x4 ev = *(const f32x4*)&eps[c];
    f32x4 hv = *(const f32x4*)&nh[v * 4];
    f32x4 zv = *(const f32x4*)&nz[v * 4];
    *(f32x4*)&xn[v * 4] = (1.0f + ev) * hv + zv;
  }
}

__global__ void dice_combine_edge(const float* __restrict__ eh, const float* __restrict__ nz,
                                  const int* __restrict__ src, const int* __restrict__ dst,
                                  const float* __restrict__ eps, float* __restrict__ xe, int E) {
  size_t total = (size_t)E * 32;
  size_t stride = (size_t)gridDim.x * blockDim.x;
  for (size_t v = (size_t)blockIdx.x * blockDim.x + threadIdx.x; v < total; v += stride) {
    int c = (int)(v & 31) * 4;
    size_t e = v >> 5;
    f32x4 ev = *(const f32x4*)&eps[c];
    f32x4 hv = *(const f32x4*)&eh[v * 4];
    f32x4 zs = *(const f32x4*)(nz + (size_t)src[e] * HID + c);
    f32x4 zd = *(const f32x4*)(nz + (size_t)dst[e] * HID + c);
    *(f32x4*)&xe[v * 4] = (1.0f + ev) * hv + zs - zd;
  }
}

__global__ void dice_copy4(const float* __restrict__ a, float* __restrict__ b, size_t n4) {
  size_t stride = (size_t)gridDim.x * blockDim.x;
  for (size_t i = (size_t)blockIdx.x * blockDim.x + threadIdx.x; i < n4; i += stride)
    *(f32x4*)&b[i * 4] = *(const f32x4*)&a[i * 4];
}

__global__ void dice_graph_pool(const float* __restrict__ nh, const int* __restrict__ bids,
                                float* __restrict__ gh, int rows) {
  size_t total = (size_t)rows * 32;
  size_t stride = (size_t)gridDim.x * blockDim.x;
  for (size_t v = (size_t)blockIdx.x * blockDim.x + threadIdx.x; v < total; v += stride) {
    int c = (int)(v & 31) * 4;
    size_t r = v >> 5;
    f32x4 xv = *(const f32x4*)&nh[v * 4];
    float* pg = gh + (size_t)bids[r] * HID + c;
#pragma unroll
    for (int j = 0; j < 4; ++j) atomicAdd(&pg[j], xv[j]);
  }
}

// ---------------------------------------------------------------------------
// Host orchestration
// ---------------------------------------------------------------------------
// Per-branch parameter leaves, params dict insertion order (n, e, g branches):
enum {
  L_NF_W1 = 0, L_NF_B1, L_NF_W2, L_NF_B2,
  L_EF_W1, L_EF_B1, L_EF_W2, L_EF_B2,
  L_NH_BN_G, L_NH_BN_B, L_EH_BN_G, L_EH_BN_B,
  L_G_NF_W1, L_G_NF_B1, L_G_NF_W2, L_G_NF_B2,
  L_G_EF_W1, L_G_EF_B1, L_G_EF_W2, L_G_EF_B2,
  L_NF_EPS, L_EF_EPS,
  L_G_NF_BN_G, L_G_NF_BN_B, L_G_EF_BN_G, L_G_EF_BN_B,
  LEAVES_PER_BRANCH
};

extern "C" void kernel_launch(void* const* d_in, const int* in_sizes, int n_in,
                              void* d_out, int out_size, void* d_ws, size_t ws_size,
                              hipStream_t stream) {
  (void)in_sizes; (void)n_in; (void)out_size; (void)ws_size;
  const int N = N_NODESC, E = N_EDGESC;

  const float* nf   = (const float*)d_in[0];
  const float* ef   = (const float*)d_in[1];
  const int*   src  = (const int*)d_in[2];
  const int*   dst  = src + E;
  const int*   bids = (const int*)d_in[3];
  auto P = [&](int br, int leaf) -> const float* {
    return (const float*)d_in[4 + br * LEAVES_PER_BRANCH + leaf];
  };

  // workspace layout (floats)
  float* w = (float*)d_ws;
  size_t off = 0;
  float* nh[3]; float* eh[3];
  for (int b = 0; b < 3; ++b) { nh[b] = w + off; off += (size_t)N * HID; }
  for (int b = 0; b < 3; ++b) { eh[b] = w + off; off += (size_t)E * HID; }
  float* xn    = w + off; off += (size_t)N * HID;
  float* xe    = w + off; off += (size_t)E * HID;
  float* nz    = w + off; off += (size_t)N * HID;
  float* logit = w + off; off += E;
  float* exb   = w + off; off += E;
  float* mx    = w + off; off += N;
  float* ssum  = w + off; off += N;
  float* bnsum = w + off; off += 2 * HID;
  float* bnsc  = w + off; off += HID;
  float* bnsh  = w + off; off += HID;

  const int EB = 4096, TB = 256;
  auto fill = [&](float* p, size_t n, float v) {
    dice_fill4<<<EB, TB, 0, stream>>>(p, n / 4, v);
  };
  auto bn = [&](float* x, int rows, const float* g, const float* b) {
    fill(bnsum, 2 * HID, 0.0f);
    dice_bn_stats<<<EB, TB, 0, stream>>>(x, rows, bnsum);
    dice_bn_finalize<<<1, HID, 0, stream>>>(bnsum, g, b, 1.0f / (float)rows, bnsc, bnsh);
    dice_bn_apply<<<EB, TB, 0, stream>>>(x, rows, bnsc, bnsh);
  };
  auto gemm = [&](const float* x, int rows, const float* W, const float* b, float* y, int act) {
    dice_wmma_gemm128<<<(rows + 127) / 128, 256, 0, stream>>>(x, rows, W, b, y, act);
  };
  auto mlp128 = [&](float* x, int rows, const float* W1, const float* b1,
                    const float* W2, const float* b2, float* y) {
    gemm(x, rows, W1, b1, x, 1);    // in-place GELU hidden (block reads tile to LDS first)
    gemm(x, rows, W2, b2, y, 0);
  };

  // ---- input embeddings ----
  for (int b = 0; b < 3; ++b) {
    dice_embed_l1<<<EB, TB, 0, stream>>>(nf, N, 9, P(b, L_NF_W1), P(b, L_NF_B1), xn);
    gemm(xn, N, P(b, L_NF_W2), P(b, L_NF_B2), nh[b], 0);
    bn(nh[b], N, P(b, L_NH_BN_G), P(b, L_NH_BN_B));
    dice_embed_l1<<<EB, TB, 0, stream>>>(ef, E, 5, P(b, L_EF_W1), P(b, L_EF_B1), xe);
    gemm(xe, E, P(b, L_EF_W2), P(b, L_EF_B2), eh[b], 0);
    bn(eh[b], E, P(b, L_EH_BN_G), P(b, L_EH_BN_B));
  }

  // ---- GIN layers ----
  for (int layer = 0; layer < DEPTH; ++layer) {
    for (int b = 0; b < 3; ++b) {
      dice_attn_logit<<<8192, TB, 0, stream>>>(nh[b], eh[b], src, dst, logit, E);
      fill(mx, N, -INFINITY);
      fill(ssum, N, 0.0f);
      fill(nz, (size_t)N * HID, 0.0f);
      dice_seg_max<<<EB, TB, 0, stream>>>(logit, dst, mx, E);
      dice_seg_expsum<<<EB, TB, 0, stream>>>(logit, dst, mx, exb, ssum, E);
      dice_nz_scatter<<<8192, TB, 0, stream>>>(nh[b], src, dst, exb, ssum, nz, E);
      dice_combine_node<<<EB, TB, 0, stream>>>(nh[b], nz, P(b, L_NF_EPS), xn, N);
      dice_combine_edge<<<EB, TB, 0, stream>>>(eh[b], nz, src, dst, P(b, L_EF_EPS), xe, E);
      mlp128(xn, N, P(b, L_G_NF_W1), P(b, L_G_NF_B1), P(b, L_G_NF_W2), P(b, L_G_NF_B2), nh[b]);
      bn(nh[b], N, P(b, L_G_NF_BN_G), P(b, L_G_NF_BN_B));
      mlp128(xe, E, P(b, L_G_EF_W1), P(b, L_G_EF_B1), P(b, L_G_EF_W2), P(b, L_G_EF_B2), eh[b]);
      bn(eh[b], E, P(b, L_G_EF_BN_G), P(b, L_G_EF_BN_B));
    }
  }

  // ---- outputs: (n-branch nodes, e-branch edges, g-branch graph pool) ----
  float* out = (float*)d_out;
  dice_copy4<<<EB, TB, 0, stream>>>(nh[0], out, (size_t)N * 32);
  dice_copy4<<<EB, TB, 0, stream>>>(eh[1], out + (size_t)N * HID, (size_t)E * 32);
  float* gh = out + (size_t)N * HID + (size_t)E * HID;
  fill(gh, (size_t)N_GRAPHS * HID, 0.0f);
  dice_graph_pool<<<EB, TB, 0, stream>>>(nh[2], bids, gh, N);
}
